// Attention_Actor_66975720013858
// MI455X (gfx1250) — matrix operand order repaired and checked by reading, not compiled, not run on hardware
//
#include <hip/hip_runtime.h>
#include <hip/hip_bf16.h>

#define NA 8
#define BQ 8192
#define H  128
#define LPAD 8

typedef _Float16 half_t;
typedef __attribute__((ext_vector_type(16))) _Float16 v16h;
typedef __attribute__((ext_vector_type(8)))  _Float16 v8h;
typedef __attribute__((ext_vector_type(8)))  float    v8f;
typedef __attribute__((ext_vector_type(4)))  int      v4i;

union V16 { v16h v; v8h h[2]; };

#if __has_builtin(__builtin_amdgcn_global_load_async_to_lds_b128) && \
    __has_builtin(__builtin_amdgcn_s_wait_asynccnt)
#define HAS_ASYNC 1
typedef __attribute__((address_space(1))) v4i* gp_v4i;   // global int4*
typedef __attribute__((address_space(3))) v4i* lp_v4i;   // LDS int4*
#else
#define HAS_ASYNC 0
#endif

__device__ __forceinline__ v8f wmma32(v16h a, v16h b, v8f c) {
  return __builtin_amdgcn_wmma_f32_16x16x32_f16(false, a, false, b, (short)0, c, false, false);
}
__device__ __forceinline__ float lrelu(float x) { return x >= 0.f ? x : 0.01f * x; }

// Stage a 128x128 fp16 matrix (row-major in global) into padded LDS [128][128+LPAD].
// Uses gfx1250 async global->LDS copies when available (ASYNCcnt), else VGPR copies.
__device__ __forceinline__ void stage_weights(half_t (*W)[H + LPAD],
                                              const half_t* __restrict__ g) {
  int t = threadIdx.x;                // 256 threads, 2048 16B chunks
#if HAS_ASYNC
#pragma unroll
  for (int k = 0; k < 8; k++) {
    int c = k * 256 + t;              // lane-consecutive chunks -> coalesced
    int r = c >> 4, col = (c & 15) * 8;
    __builtin_amdgcn_global_load_async_to_lds_b128(
        (gp_v4i)(g + (size_t)r * H + col), (lp_v4i)&W[r][col], 0, 0);
  }
  __builtin_amdgcn_s_wait_asynccnt(0);
#else
#pragma unroll
  for (int k = 0; k < 8; k++) {
    int c = k * 256 + t;
    int r = c >> 4, col = (c & 15) * 8;
    *(v8h*)&W[r][col] = *(const v8h*)&g[(size_t)r * H + col];
  }
#endif
  __syncthreads();
}

// ---------------- K1: BN statistics (mean, rstd) per (agent, raw column) --------------
__global__ __launch_bounds__(256) void k_bn_stats(const float* __restrict__ states,
                                                  float* __restrict__ stats) {
  int a = blockIdx.x / 48, c = blockIdx.x % 48;
  const float* p = states + (size_t)a * BQ * 48 + c;
  float s = 0.f, sq = 0.f;
  for (int b = threadIdx.x; b < BQ; b += 256) {
    float x = p[(size_t)b * 48];
    s += x; sq += x * x;
  }
  __shared__ float rs[256], rq[256];
  int t = threadIdx.x;
  rs[t] = s; rq[t] = sq; __syncthreads();
  for (int k = 128; k > 0; k >>= 1) {
    if (t < k) { rs[t] += rs[t + k]; rq[t] += rq[t + k]; }
    __syncthreads();
  }
  if (t == 0) {
    float m = rs[0] * (1.0f / BQ);
    float v = rq[0] * (1.0f / BQ) - m * m;
    stats[(a * 48 + c) * 2 + 0] = m;
    stats[(a * 48 + c) * 2 + 1] = rsqrtf(v + 1e-5f);
  }
}

// ------- K2: Q = Wsel @ Wk^T (fp16 [head][k][j]) + fp16 conversion of Wv0/Wv1 ---------
__global__ __launch_bounds__(256) void k_prep(const float* __restrict__ Wsel0,
                                              const float* __restrict__ Wk0,
                                              const float* __restrict__ Wsel1,
                                              const float* __restrict__ Wk1,
                                              const float* __restrict__ Wv0,
                                              const float* __restrict__ Wv1,
                                              half_t* __restrict__ wq,
                                              half_t* __restrict__ wvh) {
  int tid = blockIdx.x * 256 + threadIdx.x;     // 0..65535
  if (tid < 32768) {                            // Q outputs
    int h = tid >> 14, k = (tid >> 7) & 127, j = tid & 127;
    const float* ws = (h ? Wsel1 : Wsel0) + k * H;
    const float* wk = (h ? Wk1 : Wk0) + j * H;
    float s = 0.f;
    for (int u = 0; u < H; u++) s += ws[u] * wk[u];
    wq[tid] = (half_t)s;
  } else {                                      // Wv fp16 copies
    int u = tid - 32768;
    int h = u >> 14, e = u & 16383;
    wvh[u] = (half_t)((h ? Wv1 : Wv0)[e]);
  }
}

// ---------------- K3: BN + small encoders -> fp16 EN / CAT0 / CAT1 --------------------
__global__ __launch_bounds__(128) void k_encode(
    const float* __restrict__ states, const float* __restrict__ stats,
    const float* __restrict__ Wen, const float* __restrict__ ben,
    const float* __restrict__ Woa, const float* __restrict__ boa,
    const float* __restrict__ Wg,  const float* __restrict__ bg,
    half_t* __restrict__ EN, half_t* __restrict__ C0, half_t* __restrict__ C1) {
  int a = blockIdx.y;
  int h = threadIdx.x;                       // output feature
  int row0 = blockIdx.x * 64;
  __shared__ float sm[48], sr[48];
  if (h < 48) { sm[h] = stats[(a * 48 + h) * 2]; sr[h] = stats[(a * 48 + h) * 2 + 1]; }
  __syncthreads();
  float wen[4], woa[4], wg[2];
#pragma unroll
  for (int f = 0; f < 4; f++) wen[f] = Wen[((size_t)a * 4 + f) * H + h];
#pragma unroll
  for (int f = 0; f < 4; f++) woa[f] = Woa[((size_t)a * 4 + f) * H + h];
#pragma unroll
  for (int f = 0; f < 2; f++) wg[f]  = Wg [((size_t)a * 2 + f) * H + h];
  float be = ben[a * H + h], bo = boa[a * H + h], bgl = bg[a * H + h];

  for (int r = 0; r < 64; r++) {
    int b = row0 + r;
    size_t m = (size_t)a * BQ + b;
    const float* x = states + m * 48;
    float acc = be;
#pragma unroll
    for (int f = 0; f < 4; f++) acc += (x[f] - sm[f]) * sr[f] * wen[f];
    EN[m * H + h] = (half_t)lrelu(acc);
#pragma unroll
    for (int i = 0; i < 7; i++) {
      float s = bo;
#pragma unroll
      for (int f = 0; f < 4; f++) {
        int c = 4 + i * 4 + f;
        s += (x[c] - sm[c]) * sr[c] * woa[f];
      }
      C0[(m * 7 + i) * H + h] = (half_t)lrelu(s);
    }
#pragma unroll
    for (int i = 0; i < 8; i++) {
      float s = bgl;
#pragma unroll
      for (int f = 0; f < 2; f++) {
        int c = 32 + i * 2 + f;
        s += (x[c] - sm[c]) * sr[c] * wg[f];
      }
      C1[(m * 8 + i) * H + h] = (half_t)lrelu(s);
    }
  }
}

// ---------------- K4: selk = (EN @ Q) * scale, WMMA, Q staged in LDS -------------------
__global__ __launch_bounds__(256) void k_selk(const half_t* __restrict__ EN,
                                              const half_t* __restrict__ WQ,
                                              half_t* __restrict__ SELK) {
  int head = blockIdx.y;
  half_t* out = SELK + (size_t)head * NA * BQ * H;
  __shared__ half_t W[H][H + LPAD];
  stage_weights(W, WQ + (size_t)head * H * H);

  int wave = threadIdx.x >> 5, lane = threadIdx.x & 31;
  int m0 = blockIdx.x * 128 + wave * 16;
  int row = lane & 15, hi = lane >> 4;
  const half_t* ab = EN + (size_t)(m0 + row) * H;
  V16 af[4];
#pragma unroll
  for (int ks = 0; ks < 4; ks++) {
    af[ks].h[0] = *(const v8h*)&ab[ks * 32 + hi * 8];
    af[ks].h[1] = *(const v8h*)&ab[ks * 32 + hi * 8 + 16];
  }
  const float scale = 0.08838834764831845f;   // 1/sqrt(128)
#pragma unroll
  for (int nt = 0; nt < 8; nt++) {
    V16 bf[4];
#pragma unroll
    for (int ks = 0; ks < 4; ks++) {          // preload all K-step B-frags
      for (int s = 0; s < 8; s++) {
        bf[ks].h[0][s] = W[ks * 32 + hi * 8 + s][nt * 16 + row];
        bf[ks].h[1][s] = W[ks * 32 + hi * 8 + 16 + s][nt * 16 + row];
      }
    }
    v8f acc = {};
#pragma unroll
    for (int ks = 0; ks < 4; ks++)            // back-to-back WMMA chain
      acc = wmma32(af[ks].v, bf[ks].v, acc);
    int col = row + nt * 16, mb = m0 + hi * 8;
#pragma unroll
    for (int r = 0; r < 8; r++)
      out[(size_t)(mb + r) * H + col] = (half_t)(acc[r] * scale);
  }
}

// ---------------- K5: logits + softmax + weighted leaky(enc@Wv+bv) -> OV ---------------
template <int I>
__global__ __launch_bounds__(256) void k_attn(const half_t* __restrict__ SELK,
                                              const half_t* __restrict__ CAT,
                                              const half_t* __restrict__ Wvh,
                                              const float* __restrict__ bv,
                                              half_t* __restrict__ OV) {
  __shared__ half_t W[H][H + LPAD];
  __shared__ float wsm[8][16][8];
  stage_weights(W, Wvh);

  int wave = threadIdx.x >> 5, lane = threadIdx.x & 31;
  int m0 = blockIdx.x * 128 + wave * 16;
  int row = lane & 15, hi = lane >> 4;
  size_t m = (size_t)m0 + row;

  // cache this lane's selk K-half (64 halves)
  v8h sk[8];
  const half_t* sp = SELK + m * H + hi * 64;
#pragma unroll
  for (int c = 0; c < 8; c++) sk[c] = *(const v8h*)&sp[c * 8];

  // logits + softmax (lane pairs split K; shfl_xor(16) completes each dot)
  float w[I];
  float mx = -1e30f;
#pragma unroll
  for (int i = 0; i < I; i++) {
    const half_t* ep = CAT + (m * I + i) * H + hi * 64;
    float p = 0.f;
#pragma unroll
    for (int c = 0; c < 8; c++) {
      v8h e = *(const v8h*)&ep[c * 8];
#pragma unroll
      for (int j = 0; j < 8; j++) p += (float)sk[c][j] * (float)e[j];
    }
    p += __shfl_xor(p, 16, 32);
    w[i] = p;                                  // selk already carries 1/sqrt(H)
    mx = fmaxf(mx, w[i]);
  }
  float sum = 0.f;
#pragma unroll
  for (int i = 0; i < I; i++) { w[i] = __expf(w[i] - mx); sum += w[i]; }
  float inv = 1.f / sum;
  if (lane < 16) {
#pragma unroll
    for (int i = 0; i < I; i++) wsm[wave][row][i] = w[i] * inv;
  }
  __syncthreads();

  float bvr[8];
#pragma unroll
  for (int nt = 0; nt < 8; nt++) bvr[nt] = bv[row + nt * 16];
  v8f ov[8];
#pragma unroll
  for (int nt = 0; nt < 8; nt++) { v8f z = {}; ov[nt] = z; }

  for (int i = 0; i < I; i++) {
    float wv[8];
#pragma unroll
    for (int r = 0; r < 8; r++) wv[r] = wsm[wave][r + 8 * hi][i];
    V16 af[4];
    const half_t* ab = CAT + (m * I + i) * H;
#pragma unroll
    for (int ks = 0; ks < 4; ks++) {
      af[ks].h[0] = *(const v8h*)&ab[ks * 32 + hi * 8];
      af[ks].h[1] = *(const v8h*)&ab[ks * 32 + hi * 8 + 16];
    }
#pragma unroll
    for (int nt = 0; nt < 8; nt++) {
      V16 bf[4];
#pragma unroll
      for (int ks = 0; ks < 4; ks++) {
        for (int s = 0; s < 8; s++) {
          bf[ks].h[0][s] = W[ks * 32 + hi * 8 + s][nt * 16 + row];
          bf[ks].h[1][s] = W[ks * 32 + hi * 8 + 16 + s][nt * 16 + row];
        }
      }
      v8f acc = {};
#pragma unroll
      for (int ks = 0; ks < 4; ks++)
        acc = wmma32(af[ks].v, bf[ks].v, acc);
#pragma unroll
      for (int r = 0; r < 8; r++)
        ov[nt][r] += wv[r] * lrelu(acc[r] + bvr[nt]);
    }
  }
#pragma unroll
  for (int nt = 0; nt < 8; nt++) {
    int col = row + nt * 16, mb = m0 + 8 * hi;
#pragma unroll
    for (int r = 0; r < 8; r++)
      OV[(size_t)(mb + r) * H + col] = (half_t)ov[nt][r];
  }
}

// ---------------- K6: per-agent merge [en|ov0|ov1] @ Wm + bm, tanh ---------------------
__global__ __launch_bounds__(256) void k_merge(const half_t* __restrict__ EN,
                                               const half_t* __restrict__ OV0,
                                               const half_t* __restrict__ OV1,
                                               const float* __restrict__ Wm,
                                               const float* __restrict__ bm,
                                               float* __restrict__ out) {
  int m = blockIdx.x * 256 + threadIdx.x;     // 0..65535
  int a = m >> 13;
  const float* wm = Wm + (size_t)a * 384 * 2;
  float s0 = bm[a * 2], s1 = bm[a * 2 + 1];
  const v8h* e  = (const v8h*)(EN  + (size_t)m * H);
  const v8h* o0 = (const v8h*)(OV0 + (size_t)m * H);
  const v8h* o1 = (const v8h*)(OV1 + (size_t)m * H);
#pragma unroll 4
  for (int c = 0; c < 16; c++) {
    v8h x = e[c];
#pragma unroll
    for (int j = 0; j < 8; j++) {
      float v = (float)x[j];
      s0 += v * wm[(c * 8 + j) * 2]; s1 += v * wm[(c * 8 + j) * 2 + 1];
    }
  }
#pragma unroll 4
  for (int c = 0; c < 16; c++) {
    v8h x = o0[c];
#pragma unroll
    for (int j = 0; j < 8; j++) {
      float v = (float)x[j];
      s0 += v * wm[(128 + c * 8 + j) * 2]; s1 += v * wm[(128 + c * 8 + j) * 2 + 1];
    }
  }
#pragma unroll 4
  for (int c = 0; c < 16; c++) {
    v8h x = o1[c];
#pragma unroll
    for (int j = 0; j < 8; j++) {
      float v = (float)x[j];
      s0 += v * wm[(256 + c * 8 + j) * 2]; s1 += v * wm[(256 + c * 8 + j) * 2 + 1];
    }
  }
  out[m * 2 + 0] = tanhf(s0);
  out[m * 2 + 1] = tanhf(s1);
}

extern "C" void kernel_launch(void* const* d_in, const int* in_sizes, int n_in,
                              void* d_out, int out_size, void* d_ws, size_t ws_size,
                              hipStream_t stream) {
  const float* states = (const float*)d_in[0];
  const float* Wen  = (const float*)d_in[1];
  const float* ben  = (const float*)d_in[2];
  const float* Woa  = (const float*)d_in[3];
  const float* boa  = (const float*)d_in[4];
  const float* Wg   = (const float*)d_in[5];
  const float* bg   = (const float*)d_in[6];
  const float* Wk0  = (const float*)d_in[7];
  const float* Wsel0= (const float*)d_in[8];
  const float* Wv0  = (const float*)d_in[9];
  const float* bv0  = (const float*)d_in[10];
  const float* Wk1  = (const float*)d_in[11];
  const float* Wsel1= (const float*)d_in[12];
  const float* Wv1  = (const float*)d_in[13];
  const float* bv1  = (const float*)d_in[14];
  const float* Wm   = (const float*)d_in[15];
  const float* bm   = (const float*)d_in[16];

  char* ws = (char*)d_ws;
  const size_t M = (size_t)NA * BQ;                   // 65536 rows
  float*  stats = (float*)(ws + 0);                   // 8*48*2 f32
  half_t* WQ    = (half_t*)(ws + 4096);               // 2*128*128 f16
  half_t* WVH   = WQ + 2 * H * H;                     // 2*128*128 f16
  half_t* EN    = (half_t*)(ws + 139264);             // M*128 f16
  half_t* C0    = EN + M * H;                         // M*7*128
  half_t* C1    = C0 + M * 7 * H;                     // M*8*128
  half_t* SELK  = C1 + M * 8 * H;                     // 2*M*128
  half_t* OV0   = SELK + 2 * M * H;                   // M*128
  half_t* OV1   = OV0 + M * H;                        // M*128

  k_bn_stats<<<NA * 48, 256, 0, stream>>>(states, stats);
  k_prep<<<256, 256, 0, stream>>>(Wsel0, Wk0, Wsel1, Wk1, Wv0, Wv1, WQ, WVH);
  k_encode<<<dim3(BQ / 64, NA), 128, 0, stream>>>(states, stats, Wen, ben, Woa, boa,
                                                  Wg, bg, EN, C0, C1);
  k_selk<<<dim3(M / 128, 2), 256, 0, stream>>>(EN, WQ, SELK);
  k_attn<7><<<M / 128, 256, 0, stream>>>(SELK,         C0, WVH,         bv0, OV0);
  k_attn<8><<<M / 128, 256, 0, stream>>>(SELK + M * H, C1, WVH + H * H, bv1, OV1);
  k_merge<<<M / 256, 256, 0, stream>>>(EN, OV0, OV1, Wm, bm, (float*)d_out);
}
